// PSPModule_24412594111128
// MI455X (gfx1250) — compile-verified, matrix operand order checked by reading, not run on hardware
//
#include <hip/hip_runtime.h>
#include <hip/hip_bf16.h>

typedef __bf16 bf16;
typedef __attribute__((ext_vector_type(16))) __bf16 v16bf;
typedef __attribute__((ext_vector_type(8)))  float  v8f;

#define EPS   1e-5f
#define SLOPE 0.01f

#define NB    2
#define CIN   2048
#define HH    64
#define WW    64
#define COUT  512
#define CCAT  4096
#define KTOT  (CCAT * 9)   // 36864

// -------- workspace layout (bytes) --------
#define OFF_ABF   ((size_t)0)           // bf16 [512][36864]           = 37,748,736
#define OFF_CAT   ((size_t)37748736)    // bf16 [2][64][64][4096] NHWC = 67,108,864
#define OFF_POOL  ((size_t)104857600)   // f32  [100][2048]            =    819,200
#define OFF_TMP   ((size_t)105676800)   // f32  [51200]                =    204,800
#define OFF_PRIOR ((size_t)105881600)   // f32  [51200]                =    204,800
#define OFF_STATS ((size_t)106086400)   // f32  [512][2]               =      4,096

// LDS geometry for the WMMA conv
#define BS_STRIDE 80                    // bytes per pixel row (32 bf16 + pad, 16B aligned)
#define BS_ROW    (66 * BS_STRIDE)      // 5280 B  (66 pixels incl. halo)
#define BS_BUF    (3 * BS_ROW)          // 15840 B (3 halo rows)
#define AS_BUF    (128 * BS_STRIDE)     // 10240 B (128 out-ch x 32 k)

// ---- CDNA5 async global->LDS copy (ASYNCcnt-tracked, bypasses VGPRs) ------
#if defined(__gfx1250__) && __has_builtin(__builtin_amdgcn_global_load_async_to_lds_b128) && \
    __has_builtin(__builtin_amdgcn_s_wait_asynccnt)
#define HAVE_ASYNC_LDS 1
#else
#define HAVE_ASYNC_LDS 0
#endif

typedef int v4i_t __attribute__((vector_size(16)));
typedef __attribute__((address_space(1))) v4i_t* gbl_v4i_p;   // global int4*
typedef __attribute__((address_space(3))) v4i_t* lds_v4i_p;   // LDS int4*

__device__ __forceinline__ void async_b128(const void* g, void* l) {
#if HAVE_ASYNC_LDS
  __builtin_amdgcn_global_load_async_to_lds_b128((gbl_v4i_p)g, (lds_v4i_p)l, 0, 0);
#else
  *(uint4*)l = *(const uint4*)g;
#endif
}
__device__ __forceinline__ void wait_async_all() {
#if HAVE_ASYNC_LDS
  __builtin_amdgcn_s_wait_asynccnt(0);
#endif
}

// ---------------------------------------------------------------------------
// K0: repack 3x3 weights [O][I][3][3] -> bf16 A[O][tap*4096 + c]
// ---------------------------------------------------------------------------
__global__ void k_cvt_w(const float* __restrict__ wb, bf16* __restrict__ abf) {
  int t = blockIdx.x * 256 + threadIdx.x;
  if (t >= COUT * KTOT) return;
  int o = t / KTOT;
  int r = t % KTOT;
  int tap = r / CCAT;
  int c   = r % CCAT;
  abf[t] = (bf16)wb[(o * CCAT + c) * 9 + tap];
}

// ---------------------------------------------------------------------------
// K1: adaptive average pool (torch floor/ceil windows) -> pooled[win][c]
// win enumeration: s=1 -> [0,2), s=2 -> [2,10), s=3 -> [10,28), s=6 -> [28,100)
// ---------------------------------------------------------------------------
__global__ void k_pool(const float* __restrict__ feats, float* __restrict__ pooled) {
  int w  = blockIdx.x;          // 0..99
  int cg = blockIdx.y;          // 0..31  (64 channels each)
  int s, r;
  if (w < 2)       { s = 1; r = w;      }
  else if (w < 10) { s = 2; r = w - 2;  }
  else if (w < 28) { s = 3; r = w - 10; }
  else             { s = 6; r = w - 28; }
  int ss = s * s;
  int n = r / ss, q = r % ss, wy = q / s, wx = q % s;
  int hs = wy * HH / s, he = ((wy + 1) * HH + s - 1) / s;
  int xs = wx * WW / s, xe = ((wx + 1) * WW + s - 1) / s;
  int rows = he - hs, cols = xe - xs, area = rows * cols;

  __shared__ float red[256];
  int tid = threadIdx.x;
  for (int ch = 0; ch < 64; ++ch) {
    int c = cg * 64 + ch;
    const float* base = feats + ((size_t)(n * CIN + c)) * HH * WW;
    float acc = 0.f;
    for (int p = tid; p < area; p += 256) {
      int yy = hs + p / cols, xx = xs + p % cols;
      acc += base[yy * WW + xx];
    }
    red[tid] = acc;
    __syncthreads();
    for (int st = 128; st > 0; st >>= 1) {
      if (tid < st) red[tid] += red[tid + st];
      __syncthreads();
    }
    if (tid == 0) pooled[(size_t)w * CIN + c] = red[0] / (float)area;
    __syncthreads();
  }
}

// ---------------------------------------------------------------------------
// K2a: 1x1 conv on pooled tensors -> tmp[t], t = [size ranges][o][n*ss+win]
// ---------------------------------------------------------------------------
__global__ void k_c1(const float* __restrict__ w0, const float* __restrict__ w1,
                     const float* __restrict__ w2, const float* __restrict__ w3,
                     const float* __restrict__ pooled, float* __restrict__ tmp) {
  int t = blockIdx.x * 256 + threadIdx.x;    // 0..51199
  if (t >= 51200) return;
  int s, r, woff;
  const float* w;
  if (t < 1024)       { s = 1; r = t;         w = w0; woff = 0;  }
  else if (t < 5120)  { s = 2; r = t - 1024;  w = w1; woff = 2;  }
  else if (t < 14336) { s = 3; r = t - 5120;  w = w2; woff = 10; }
  else                { s = 6; r = t - 14336; w = w3; woff = 28; }
  int ss = s * s, nw = 2 * ss;
  int o = r / nw, nwin = r % nw;
  const float* wr = w + (size_t)o * CIN;
  const float* pr = pooled + (size_t)(woff + nwin) * CIN;
  float acc = 0.f;
  for (int c = 0; c < CIN; ++c) acc += wr[c] * pr[c];
  tmp[t] = acc;
}

// ---------------------------------------------------------------------------
// K2b: per-(size,channel) batch-norm (biased var over n,win) + leaky relu
// ---------------------------------------------------------------------------
__global__ void k_bn1(const float* __restrict__ tmp,
                      const float* g0, const float* b0, const float* g1, const float* b1,
                      const float* g2, const float* b2, const float* g3, const float* b3,
                      float* __restrict__ prior) {
  int t = blockIdx.x * blockDim.x + threadIdx.x;   // sIdx*512 + o
  if (t >= 4 * COUT) return;
  int sIdx = t >> 9, o = t & 511;
  int s = (sIdx == 0 ? 1 : sIdx == 1 ? 2 : sIdx == 2 ? 3 : 6);
  int ss = s * s, nw = 2 * ss;
  size_t base = (size_t)(sIdx == 0 ? 0 : sIdx == 1 ? 1024 : sIdx == 2 ? 5120 : 14336)
              + (size_t)o * nw;
  float mean = 0.f;
  for (int i = 0; i < nw; ++i) mean += tmp[base + i];
  mean /= (float)nw;
  float m2 = 0.f;
  for (int i = 0; i < nw; ++i) { float d = tmp[base + i] - mean; m2 += d * d; }
  float var = m2 / (float)nw;
  const float* g = (sIdx == 0 ? g0 : sIdx == 1 ? g1 : sIdx == 2 ? g2 : g3);
  const float* b = (sIdx == 0 ? b0 : sIdx == 1 ? b1 : sIdx == 2 ? b2 : b3);
  float sc = rsqrtf(var + EPS) * g[o], sh = b[o];
  for (int i = 0; i < nw; ++i) {
    float y = (tmp[base + i] - mean) * sc + sh;
    prior[base + i] = (y >= 0.f) ? y : SLOPE * y;
  }
}

// ---------------------------------------------------------------------------
// K3a: align-corners bilinear upsample of priors -> NHWC catT channels [0,2048)
// ---------------------------------------------------------------------------
__global__ void k_cat_prior(const float* __restrict__ prior, bf16* __restrict__ catT) {
  size_t i = (size_t)blockIdx.x * 256 + threadIdx.x;   // NB*64*64*2048
  if (i >= (size_t)NB * HH * WW * 2048) return;
  int c = (int)(i & 2047);
  int x = (int)((i >> 11) & 63);
  int y = (int)((i >> 17) & 63);
  int n = (int)(i >> 23);
  int sIdx = c >> 9, o = c & 511;
  int s = (sIdx == 0 ? 1 : sIdx == 1 ? 2 : sIdx == 2 ? 3 : 6);
  int ss = s * s;
  size_t so = (size_t)(sIdx == 0 ? 0 : sIdx == 1 ? 1024 : sIdx == 2 ? 5120 : 14336);
  const float* P = prior + so + (size_t)o * (2 * ss) + (size_t)n * ss;
  float v;
  if (s == 1) {
    v = P[0];
  } else {
    float sc = (float)(s - 1) / 63.f;
    float yc = y * sc, xc = x * sc;
    int y0 = (int)yc, x0 = (int)xc;
    int y1 = min(y0 + 1, s - 1), x1 = min(x0 + 1, s - 1);
    float fy = yc - (float)y0, fx = xc - (float)x0;
    float a = P[y0 * s + x0], b = P[y0 * s + x1];
    float cc = P[y1 * s + x0], d = P[y1 * s + x1];
    v = (a * (1.f - fx) + b * fx) * (1.f - fy) + (cc * (1.f - fx) + d * fx) * fy;
  }
  catT[(((size_t)n * HH + y) * WW + x) * CCAT + c] = (bf16)v;
}

// ---------------------------------------------------------------------------
// K3b: LDS tile-transpose of feats (NCHW f32) -> NHWC catT channels [2048,4096)
// ---------------------------------------------------------------------------
__global__ __launch_bounds__(256) void k_cat_feats(const float* __restrict__ feats,
                                                   bf16* __restrict__ catT) {
  int rowb = blockIdx.x;          // n*64 + y
  int cb   = blockIdx.y * 32;     // channel base within [0,2048)
  int n = rowb >> 6, y = rowb & 63;
  __shared__ __align__(16) bf16 tileT[64 * 40];   // x-major, 80B stride
  int tid = threadIdx.x;
  for (int e = tid; e < 2048; e += 256) {
    int ch = e >> 6, x = e & 63;
    float v = feats[(((size_t)n * CIN + (cb + ch)) << 12) + (y << 6) + x];
    tileT[x * 40 + ch] = (bf16)v;
  }
  __syncthreads();
  int x = tid >> 2, q = tid & 3;
  uint4 v = *(const uint4*)((const unsigned char*)tileT + x * 80 + q * 16);
  *(uint4*)((unsigned char*)catT +
            ((((size_t)n * HH + y) * WW + x) * (size_t)CCAT + 2048 + cb) * 2 + q * 16) = v;
}

// ---------------------------------------------------------------------------
// K4: 3x3 conv as implicit GEMM with v_wmma_f32_16x16x32_bf16.
// Block (256 thr = 8 wave32): one image row (n,y) x 128 out channels.
// Wave (wm 0..3, wn 0..1): 32 out-ch x 32 px = 2x2 WMMA tiles.
// Software-pipelined K loop (1152 steps = 128 chunks x 9 taps):
//   ASYNC global->LDS staging (GLOBAL_LOAD_ASYNC_TO_LDS_B128, ASYNCcnt) of
//   step t+1 overlapped with WMMAs of step t; s_wait_asynccnt 0 + barrier at
//   end of step. Halo zeros written once (persist across chunks).
// ---------------------------------------------------------------------------
__global__ __launch_bounds__(256) void k_conv3(const bf16* __restrict__ abf,
                                               const bf16* __restrict__ catT,
                                               float* __restrict__ out) {
  const int row = blockIdx.x;           // 0..127  (n*64 + y)
  const int mt  = blockIdx.y;           // 0..3
  const int n = row >> 6, y = row & 63;
  const int obase = mt * 128;
  const int tid  = threadIdx.x;
  const int wave = tid >> 5, lane = tid & 31;
  const int wm = wave >> 1, wn = wave & 1;
  const int Ln = lane & 15, hi = lane >> 4;

  __shared__ __align__(16) unsigned char sm[2 * BS_BUF + 2 * AS_BUF];   // 52160 B
  unsigned char* BsB = sm;
  unsigned char* AsB = sm + 2 * BS_BUF;

  // stage in-bounds activations (halo slots stay zero from the prologue fill)
  auto stage_B = [&](int c1, int bb) {
    unsigned char* dst = BsB + bb * BS_BUF;
    for (int e = tid; e < 3 * 66 * 4; e += 256) {
      int rr  = e / (66 * 4);
      int rem = e % (66 * 4);
      int px = rem >> 2, q = rem & 3;
      int yy = y + rr - 1, xx = px - 1;
      if (yy >= 0 && yy < HH && xx >= 0 && xx < WW) {
        const unsigned char* g = (const unsigned char*)catT +
            ((((size_t)n * HH + yy) * WW + xx) * (size_t)CCAT + c1) * 2 + q * 16;
        async_b128(g, dst + rr * BS_ROW + px * BS_STRIDE + q * 16);
      }
    }
    // prefetch next chunk's activation lines (global_prefetch)
    if (c1 + 32 < CCAT && tid < 3 * 66) {
      int rr = tid / 66, px = tid % 66;
      int yy = y + rr - 1, xx = px - 1;
      if (yy >= 0 && yy < HH && xx >= 0 && xx < WW)
        __builtin_prefetch((const unsigned char*)catT +
            ((((size_t)n * HH + yy) * WW + xx) * (size_t)CCAT + c1 + 32) * 2, 0, 3);
    }
  };
  // stage 128 out-ch x 32 k of weights for one tap
  auto stage_A = [&](int c1, int tap1, int bb) {
    unsigned char* dst = AsB + bb * AS_BUF;
    const unsigned char* gsrc =
        (const unsigned char*)abf + ((size_t)tap1 * CCAT + (size_t)c1) * 2;
    for (int e = tid; e < 512; e += 256) {
      int ol = e >> 2, q = e & 3;
      async_b128(gsrc + (size_t)(obase + ol) * (size_t)(KTOT * 2) + q * 16,
                 dst + ol * BS_STRIDE + q * 16);
    }
    // prefetch the next tap's weight rows
    if (tid < 128) {
      int tap2 = tap1 + 1, c2 = c1;
      if (tap2 == 9) { tap2 = 0; c2 = c1 + 32; }
      if (c2 < CCAT)
        __builtin_prefetch((const unsigned char*)abf +
            ((size_t)tap2 * CCAT + (size_t)c2) * 2 +
            (size_t)(obase + tid) * (size_t)(KTOT * 2), 0, 3);
    }
  };

  v8f acc[2][2] = {};

  // prologue: zero both B buffers once (persistent halo zeros), stage step 0
  {
    uint4 z = {0u, 0u, 0u, 0u};
    for (int e = tid; e < (2 * BS_BUF) / 16; e += 256)
      *(uint4*)(BsB + e * 16) = z;
  }
  __syncthreads();
  stage_B(0, 0);
  stage_A(0, 0, 0);
  wait_async_all();
  __syncthreads();

  for (int k = 0; k < CCAT / 32; ++k) {
    const int c0 = k * 32;
#pragma unroll
    for (int tap = 0; tap < 9; ++tap) {
      // ---- issue async staging for the next step (overlaps WMMA below)
      if (tap < 8) {
        stage_A(c0, tap + 1, (k + tap + 1) & 1);
      } else if (k + 1 < CCAT / 32) {
        stage_B(c0 + 32, (k + 1) & 1);
        stage_A(c0 + 32, 0, (k + 1) & 1);     // (k*9+9) & 1 == (k+1) & 1
      }
      // ---- compute current step (buffers staged last step, already synced)
      const int dy = tap / 3, dx = tap % 3;
      const unsigned char* bbase = BsB + (k & 1) * BS_BUF + dy * BS_ROW;
      const unsigned char* abase = AsB + ((k + tap) & 1) * AS_BUF;

      union Frag { v16bf v; uint4 u[2]; } a[2], b[2];
      for (int i = 0; i < 2; ++i) {
        const unsigned char* p = abase + (wm * 32 + i * 16 + Ln) * BS_STRIDE + hi * 16;
        a[i].u[0] = *(const uint4*)p;
        a[i].u[1] = *(const uint4*)(p + 32);
      }
      for (int j = 0; j < 2; ++j) {
        const unsigned char* p = bbase + (wn * 32 + j * 16 + Ln + dx) * BS_STRIDE + hi * 16;
        b[j].u[0] = *(const uint4*)p;
        b[j].u[1] = *(const uint4*)(p + 32);
      }
      for (int i = 0; i < 2; ++i)
        for (int j = 0; j < 2; ++j)
          acc[i][j] = __builtin_amdgcn_wmma_f32_16x16x32_bf16(
              false, a[i].v, false, b[j].v, (short)0, acc[i][j], false, false);
      // ---- my async copies for step t+1 must land before anyone reads them
      wait_async_all();
      __syncthreads();
    }
  }

  // ---- epilogue: C/D layout -> raw conv output in d_out (pre-BN, f32, NCHW)
  for (int i = 0; i < 2; ++i)
    for (int j = 0; j < 2; ++j) {
      int xo = wn * 32 + j * 16 + Ln;
      for (int rg = 0; rg < 8; ++rg) {
        int o = obase + wm * 32 + i * 16 + rg + hi * 8;
        out[(((size_t)n * COUT + o) << 12) + (y << 6) + xo] = acc[i][j][rg];
      }
    }
}

// ---------------------------------------------------------------------------
// K5: per-channel mean / biased variance over (N,H,W) of raw conv output
// ---------------------------------------------------------------------------
__global__ void k_stats(const float* __restrict__ out, float* __restrict__ stats) {
  int o = blockIdx.x;
  __shared__ float s1[256], s2[256];
  int tid = threadIdx.x;
  float a = 0.f, b = 0.f;
  for (int i = tid; i < NB * HH * WW; i += 256) {
    int nn = i >> 12, sp = i & 4095;
    float v = out[(((size_t)nn * COUT + o) << 12) + sp];
    a += v; b += v * v;
  }
  s1[tid] = a; s2[tid] = b;
  __syncthreads();
  for (int st = 128; st > 0; st >>= 1) {
    if (tid < st) { s1[tid] += s1[tid + st]; s2[tid] += s2[tid + st]; }
    __syncthreads();
  }
  if (tid == 0) {
    float m = s1[0] / 8192.f;
    stats[o * 2]     = m;
    stats[o * 2 + 1] = s2[0] / 8192.f - m * m;
  }
}

// ---------------------------------------------------------------------------
// K6: in-place BN + leaky relu on d_out
// ---------------------------------------------------------------------------
__global__ void k_bnapply(float* __restrict__ out, const float* __restrict__ stats,
                          const float* __restrict__ g, const float* __restrict__ bta) {
  size_t i = (size_t)blockIdx.x * 256 + threadIdx.x;
  if (i >= (size_t)NB * COUT * HH * WW) return;
  int o = (int)((i >> 12) & 511);
  float m = stats[o * 2], v = stats[o * 2 + 1];
  float val = (out[i] - m) * rsqrtf(v + EPS) * g[o] + bta[o];
  out[i] = (val >= 0.f) ? val : SLOPE * val;
}

// ---------------------------------------------------------------------------
extern "C" void kernel_launch(void* const* d_in, const int* in_sizes, int n_in,
                              void* d_out, int out_size, void* d_ws, size_t ws_size,
                              hipStream_t stream) {
  (void)in_sizes; (void)n_in; (void)out_size; (void)ws_size;
  const float* feats = (const float*)d_in[0];
  const float* w0 = (const float*)d_in[1];
  const float* g0 = (const float*)d_in[2];
  const float* b0 = (const float*)d_in[3];
  const float* w1 = (const float*)d_in[4];
  const float* g1 = (const float*)d_in[5];
  const float* b1 = (const float*)d_in[6];
  const float* w2 = (const float*)d_in[7];
  const float* g2 = (const float*)d_in[8];
  const float* b2 = (const float*)d_in[9];
  const float* w3 = (const float*)d_in[10];
  const float* g3 = (const float*)d_in[11];
  const float* b3 = (const float*)d_in[12];
  const float* wb = (const float*)d_in[13];
  const float* gb = (const float*)d_in[14];
  const float* bb = (const float*)d_in[15];
  float* out = (float*)d_out;

  char* ws = (char*)d_ws;
  bf16*  abf    = (bf16*)(ws + OFF_ABF);
  bf16*  catT   = (bf16*)(ws + OFF_CAT);
  float* pooled = (float*)(ws + OFF_POOL);
  float* tmp    = (float*)(ws + OFF_TMP);
  float* prior  = (float*)(ws + OFF_PRIOR);
  float* stats  = (float*)(ws + OFF_STATS);

  k_cvt_w<<<(COUT * KTOT) / 256, 256, 0, stream>>>(wb, abf);
  k_pool<<<dim3(100, 32), 256, 0, stream>>>(feats, pooled);
  k_c1<<<200, 256, 0, stream>>>(w0, w1, w2, w3, pooled, tmp);
  k_bn1<<<8, 256, 0, stream>>>(tmp, g0, b0, g1, b1, g2, b2, g3, b3, prior);
  k_cat_prior<<<(int)(((size_t)NB * HH * WW * 2048) / 256), 256, 0, stream>>>(prior, catT);
  k_cat_feats<<<dim3(128, 64), 256, 0, stream>>>(feats, catT);
  k_conv3<<<dim3(128, 4), 256, 0, stream>>>(abf, catT, out);
  k_stats<<<COUT, 256, 0, stream>>>(out, stats);
  k_bnapply<<<(int)(((size_t)NB * COUT * HH * WW) / 256), 256, 0, stream>>>(out, stats, gb, bb);
}
